// MultiHeadSelfAttention_88484916232459
// MI455X (gfx1250) — compile-verified, matrix operand order checked by reading, not run on hardware
//
#include <hip/hip_runtime.h>

// ---------------------------------------------------------------------------
// MI455X (gfx1250, wave32) multi-head self-attention, bf16 WMMA pipeline with
// double-buffered GLOBAL_LOAD_ASYNC_TO_LDS tile staging (ASYNCcnt-tracked).
//   x[4096,1024] -> QKV GEMM -> flash attention (16 heads, Dh=64) -> out GEMM
// ---------------------------------------------------------------------------

typedef __attribute__((ext_vector_type(16))) __bf16 v16bf;
typedef __attribute__((ext_vector_type(8)))  float  v8f;

#define WMMA_BF16(A, B, C)                                                     \
  __builtin_amdgcn_wmma_f32_16x16x32_bf16(false, (A), false, (B), (short)0,    \
                                          (C), false, false)

namespace {
constexpr int kDModel = 1024;
constexpr int kHeads  = 16;
constexpr int kDHead  = 64;
constexpr int kSeq    = 2048;
constexpr int kBatch  = 2;
constexpr int kTok    = kBatch * kSeq;   // 4096
constexpr int k3C     = 3 * kDModel;     // 3072
}  // namespace

// fp32 -> bf16, round-to-nearest-even (bit-level; avoids relying on __bf16 arith)
__device__ __forceinline__ __bf16 f2bf(float f) {
  unsigned u = __builtin_bit_cast(unsigned, f);
  u += 0x7FFFu + ((u >> 16) & 1u);
  unsigned short s = (unsigned short)(u >> 16);
  return __builtin_bit_cast(__bf16, s);
}

// Generic LDS pointer -> wave-relative LDS byte address (ISA 10.2: for the LDS
// aperture, the LDS address is the low 32 bits).
__device__ __forceinline__ unsigned lds_addr(const void* p) {
  return (unsigned)(unsigned long long)p;
}

// Async memory->LDS 16B transfer, GV mode (64-bit per-lane global address).
// Tracked by ASYNCcnt; completion enforced with s_wait_asynccnt below.
__device__ __forceinline__ void async_b128(const __bf16* g, unsigned lds_byte) {
  asm volatile("global_load_async_to_lds_b128 %0, %1, off"
               :: "v"(lds_byte), "v"(g)
               : "memory");
}
__device__ __forceinline__ void wait_async_4() {
  asm volatile("s_wait_asynccnt 0x4" ::: "memory");
}
__device__ __forceinline__ void wait_async_0() {
  asm volatile("s_wait_asynccnt 0x0" ::: "memory");
}

// K-index map shared by A (16x32, lane=M) and B (32x16, lane=N) bf16 fragments.
// ISA 7.12.2: VGPR v holds elements {2v,2v+1}; lanes 0-15 = K 0..7 / 16..23,
// lanes 16-31 = K 8..15 / 24..31.
__device__ __forceinline__ int kmap(int e, int half) {
  return ((e & 8) ? (e + 8) : e) + 8 * half;
}

// Fragment from a row-major [16][ld] tile: lane r reads p[r*ld + kmap(e)].
// Serves A (rows = M) and B-transposed sources (rows = N, e.g. K in QK^T).
// Pairs (2v,2v+1) are K-consecutive -> compiler merges into ds_load_b128.
__device__ __forceinline__ v16bf load_frag_mk(const __bf16* p, int ld) {
  const int lane = (int)(threadIdx.x & 31u);
  const int r = lane & 15, h = lane >> 4;
  v16bf f;
#pragma unroll
  for (int e = 0; e < 16; ++e) f[e] = p[r * ld + kmap(e, h)];
  return f;
}

// B fragment from a row-major [K][N] tile (weight matrices, V matrix).
__device__ __forceinline__ v16bf load_frag_kn(const __bf16* p, int ld) {
  const int lane = (int)(threadIdx.x & 31u);
  const int n = lane & 15, h = lane >> 4;
  v16bf f;
#pragma unroll
  for (int e = 0; e < 16; ++e) f[e] = p[kmap(e, h) * ld + n];
  return f;
}

__device__ __forceinline__ float redmax16(float v) {
#pragma unroll
  for (int m = 1; m < 16; m <<= 1) v = fmaxf(v, __shfl_xor(v, m, 32));
  return v;
}
__device__ __forceinline__ float redsum16(float v) {
#pragma unroll
  for (int m = 1; m < 16; m <<= 1) v += __shfl_xor(v, m, 32);
  return v;
}

// ---------------------------------------------------------------------------
// fp32 -> bf16 conversion (vectorized, grid = n/1024 blocks of 256 threads)
// ---------------------------------------------------------------------------
__global__ __launch_bounds__(256) void convert_f32_bf16(
    const float* __restrict__ in, __bf16* __restrict__ out, int n) {
  int i = (int)(blockIdx.x * blockDim.x + threadIdx.x) * 4;
  if (i + 3 < n) {
    float4 v = *(const float4*)(in + i);
    out[i + 0] = f2bf(v.x);
    out[i + 1] = f2bf(v.y);
    out[i + 2] = f2bf(v.z);
    out[i + 3] = f2bf(v.w);
  } else {
    for (; i < n; ++i) out[i] = f2bf(in[i]);
  }
}

// ---------------------------------------------------------------------------
// Tiled bf16 WMMA GEMM: C[M,N] = A[M,K] * B[K,N] + bias[N]
// Block tile 128x128, BK=32; 8 waves in 2x4 grid, each wave 64x32 (4x2 tiles).
// Double-buffered async memory->LDS staging: tile t+1 streams in while tile t
// feeds the WMMAs. Sizes always divide tile dims (M%128==N%128==K%32==0).
// ---------------------------------------------------------------------------
template <bool OUT_BF16>
__global__ __launch_bounds__(256) void gemm_bf16_wmma(
    const __bf16* __restrict__ A, const __bf16* __restrict__ B,
    const float* __restrict__ bias, void* __restrict__ Cout,
    int M, int N, int K) {
  constexpr int BM = 128, BN = 128, BK = 32;
  constexpr int LDA = BK + 8;   // 40 bf16 (80B rows: 16B aligned, bank-spread)
  constexpr int LDB = BN + 8;   // 136 bf16 (272B rows)
  __shared__ __bf16 As[2][BM][LDA];
  __shared__ __bf16 Bs[2][BK][LDB];

  const int tid  = (int)threadIdx.x;
  const int wave = tid >> 5;
  const int wm = wave >> 2, wn = wave & 3;
  const int lane = tid & 31, ln = lane & 15, lh = lane >> 4;
  const int bm = (int)blockIdx.y * BM, bn = (int)blockIdx.x * BN;

  // 4 async b128 per thread per tile (A: 512 float4, B: 512 float4, /256 thr).
  auto issue_tile = [&](int k0, int buf) {
#pragma unroll
    for (int it = 0; it < 2; ++it) {
      int i = tid + 256 * it;
      int ra = i >> 2, ca = (i & 3) * 8;    // A rows: 4 float4 each
      async_b128(A + (size_t)(bm + ra) * K + k0 + ca,
                 lds_addr(&As[buf][ra][ca]));
      int rb = i >> 4, cb = (i & 15) * 8;   // B rows: 16 float4 each
      async_b128(B + (size_t)(k0 + rb) * N + bn + cb,
                 lds_addr(&Bs[buf][rb][cb]));
    }
  };

  const v8f vzero = {0.f, 0.f, 0.f, 0.f, 0.f, 0.f, 0.f, 0.f};
  v8f acc[4][2];
#pragma unroll
  for (int i = 0; i < 4; ++i)
#pragma unroll
    for (int j = 0; j < 2; ++j) acc[i][j] = vzero;

  const int nt = K / BK;
  issue_tile(0, 0);
  for (int t = 0; t < nt; ++t) {
    const int cur = t & 1;
    __syncthreads();                 // everyone done reading buf being refilled
    if (t + 1 < nt) {
      issue_tile((t + 1) * BK, cur ^ 1);
      wait_async_4();                // in-order: tile t's 4 asyncs complete
    } else {
      wait_async_0();
    }
    __syncthreads();                 // tile t visible to all waves

    v16bf af[4], bfg[2];
#pragma unroll
    for (int i = 0; i < 4; ++i)
      af[i] = load_frag_mk(&As[cur][wm * 64 + i * 16][0], LDA);
#pragma unroll
    for (int j = 0; j < 2; ++j)
      bfg[j] = load_frag_kn(&Bs[cur][0][wn * 32 + j * 16], LDB);
#pragma unroll
    for (int i = 0; i < 4; ++i)
#pragma unroll
      for (int j = 0; j < 2; ++j)
        acc[i][j] = WMMA_BF16(af[i], bfg[j], acc[i][j]);
  }

  // Epilogue: C/D layout (ISA 7.12.2): value (m,n) at vgpr r, lane n+16*(m>=8),
  // with m = r + 8*half.
#pragma unroll
  for (int i = 0; i < 4; ++i)
#pragma unroll
    for (int j = 0; j < 2; ++j) {
      const int n = bn + wn * 32 + j * 16 + ln;
      const float bv = bias[n];
#pragma unroll
      for (int r = 0; r < 8; ++r) {
        const int m = bm + wm * 64 + i * 16 + r + 8 * lh;
        const float v = acc[i][j][r] + bv;
        if (OUT_BF16)
          ((__bf16*)Cout)[(size_t)m * N + n] = f2bf(v);
        else
          ((float*)Cout)[(size_t)m * N + n] = v;
      }
    }
}

// ---------------------------------------------------------------------------
// Flash attention, bf16 WMMA. Grid = (T/128, heads, batch), 256 threads.
// Each wave owns 16 Q rows; 64-key K/V tiles stream through double-buffered
// LDS via async copies. qkv layout: [tok][3*1024], Q|K|V at cols 0|1024|2048
// (+h*64). ctx layout: [tok][1024], channel = h*64 + d.
// ---------------------------------------------------------------------------
__global__ __launch_bounds__(256) void flash_attn_wmma(
    const __bf16* __restrict__ qkv, __bf16* __restrict__ ctx) {
  constexpr int LD = kDHead + 8;  // 72 bf16 rows (144B: aligned + bank-spread)
  __shared__ __bf16 Qs[128][LD];
  __shared__ __bf16 Ks[2][64][LD];
  __shared__ __bf16 Vs[2][64][LD];
  __shared__ __bf16 Ps[8][16][LD];  // per-wave P re-layout scratch

  const int tid  = (int)threadIdx.x;
  const int wave = tid >> 5;
  const int lane = tid & 31, ln = lane & 15, lh = lane >> 4;
  const int b = (int)blockIdx.z, h = (int)blockIdx.y;
  const size_t row0 = (size_t)b * kSeq + (size_t)blockIdx.x * 128;
  const size_t srow = k3C;

  // 4 async b128 per thread per K/V tile (2 x 512 float4 / 256 threads).
  auto issue_kv = [&](int kt, int buf) {
    const size_t kbase =
        ((size_t)b * kSeq + (size_t)kt * 64) * srow + h * kDHead;
#pragma unroll
    for (int it = 0; it < 2; ++it) {
      int i = tid + 256 * it;
      int r = i >> 3, c = (i & 7) * 8;
      async_b128(qkv + kbase + (size_t)r * srow + kDModel + c,
                 lds_addr(&Ks[buf][r][c]));
      async_b128(qkv + kbase + (size_t)r * srow + 2 * kDModel + c,
                 lds_addr(&Vs[buf][r][c]));
    }
  };

  // Stage the 128x64 Q tile asynchronously, overlapped with K/V tile 0.
  {
    const __bf16* qsrc = qkv + row0 * srow + h * kDHead;
#pragma unroll
    for (int it = 0; it < 4; ++it) {    // 1024 float4, 4/thread
      int i = tid + 256 * it;
      int r = i >> 3, c = (i & 7) * 8;
      async_b128(qsrc + (size_t)r * srow + c, lds_addr(&Qs[r][c]));
    }
  }
  issue_kv(0, 0);
  wait_async_4();                       // in-order: the 4 Q asyncs are done
  __syncthreads();

  v16bf qf[2];
  qf[0] = load_frag_mk(&Qs[wave * 16][0],  LD);
  qf[1] = load_frag_mk(&Qs[wave * 16][32], LD);

  const v8f vzero = {0.f, 0.f, 0.f, 0.f, 0.f, 0.f, 0.f, 0.f};
  v8f oacc[4];
  float mrow[8], lrow[8];
#pragma unroll
  for (int j = 0; j < 4; ++j) oacc[j] = vzero;
#pragma unroll
  for (int r = 0; r < 8; ++r) { mrow[r] = -1e30f; lrow[r] = 0.f; }

  const int nkt = kSeq / 64;
  for (int kt = 0; kt < nkt; ++kt) {
    const int cur = kt & 1;
    __syncthreads();                 // readers of buf being refilled are done
    if (kt + 1 < nkt) {
      issue_kv(kt + 1, cur ^ 1);
      wait_async_4();                // K/V tile kt has landed
    } else {
      wait_async_0();
    }
    __syncthreads();

    // S = (Q K^T) * 1/sqrt(64).  K tile is [key][d] row-major == B^T source.
    v8f sacc[4];
#pragma unroll
    for (int j = 0; j < 4; ++j) {
      v8f s = vzero;
      s = WMMA_BF16(qf[0], load_frag_mk(&Ks[cur][j * 16][0],  LD), s);
      s = WMMA_BF16(qf[1], load_frag_mk(&Ks[cur][j * 16][32], LD), s);
      sacc[j] = s;
    }
#pragma unroll
    for (int j = 0; j < 4; ++j)
#pragma unroll
      for (int r = 0; r < 8; ++r) sacc[j][r] *= 0.125f;

    // Online softmax over the 64-key tile. Row m = r + 8*lh is spread across
    // the 16 lanes of this half -> shuffle-reduce within lanes 0..15 / 16..31.
    float mnew[8], sc[8], psum[8];
#pragma unroll
    for (int r = 0; r < 8; ++r) {
      float mx = fmaxf(fmaxf(sacc[0][r], sacc[1][r]),
                       fmaxf(sacc[2][r], sacc[3][r]));
      mx = redmax16(mx);
      mnew[r] = fmaxf(mrow[r], mx);
      sc[r] = __expf(mrow[r] - mnew[r]);
      mrow[r] = mnew[r];
      psum[r] = 0.f;
    }
#pragma unroll
    for (int j = 0; j < 4; ++j)
#pragma unroll
      for (int r = 0; r < 8; ++r) {
        const float p = __expf(sacc[j][r] - mnew[r]);
        psum[r] += p;
        // Re-layout P (C/D layout -> A layout) via wave-private LDS scratch;
        // same-wave DS ops are in-order, so the fragment reload below is safe.
        Ps[wave][r + 8 * lh][j * 16 + ln] = f2bf(p);
      }
#pragma unroll
    for (int r = 0; r < 8; ++r) {
      lrow[r] = lrow[r] * sc[r] + redsum16(psum[r]);
#pragma unroll
      for (int j = 0; j < 4; ++j) oacc[j][r] *= sc[r];
    }

    // O += P * V.  V tile is [key][d] row-major == [K][N] source.
#pragma unroll
    for (int kk = 0; kk < 2; ++kk) {
      v16bf pf = load_frag_mk(&Ps[wave][0][kk * 32], LD);
#pragma unroll
      for (int j = 0; j < 4; ++j)
        oacc[j] =
            WMMA_BF16(pf, load_frag_kn(&Vs[cur][kk * 32][j * 16], LD), oacc[j]);
    }
  }

#pragma unroll
  for (int j = 0; j < 4; ++j)
#pragma unroll
    for (int r = 0; r < 8; ++r) {
      const float o = oacc[j][r] / lrow[r];
      const size_t trow = row0 + wave * 16 + r + 8 * lh;
      ctx[trow * kDModel + h * kDHead + j * 16 + ln] = f2bf(o);
    }
}

// ---------------------------------------------------------------------------
// Launch: convert -> QKV GEMM -> flash attention -> output GEMM
// ---------------------------------------------------------------------------
extern "C" void kernel_launch(void* const* d_in, const int* in_sizes, int n_in,
                              void* d_out, int out_size, void* d_ws,
                              size_t ws_size, hipStream_t stream) {
  (void)in_sizes; (void)n_in; (void)out_size; (void)ws_size;

  const float* x     = (const float*)d_in[0];
  const float* qkv_w = (const float*)d_in[1];
  const float* qkv_b = (const float*)d_in[2];
  const float* out_w = (const float*)d_in[3];
  const float* out_b = (const float*)d_in[4];
  float* out = (float*)d_out;

  // Workspace carve-out (~48 MB total, all chunk sizes 256B-aligned).
  char* w = (char*)d_ws;
  __bf16* xb   = (__bf16*)w; w += (size_t)kTok * kDModel * sizeof(__bf16);
  __bf16* wqkv = (__bf16*)w; w += (size_t)kDModel * k3C * sizeof(__bf16);
  __bf16* wo   = (__bf16*)w; w += (size_t)kDModel * kDModel * sizeof(__bf16);
  __bf16* qkvb = (__bf16*)w; w += (size_t)kTok * k3C * sizeof(__bf16);
  __bf16* ctxb = (__bf16*)w; w += (size_t)kTok * kDModel * sizeof(__bf16);

  {
    int n = kTok * kDModel;     // 4,194,304
    convert_f32_bf16<<<n / 1024, 256, 0, stream>>>(x, xb, n);
  }
  {
    int n = kDModel * k3C;      // 3,145,728
    convert_f32_bf16<<<n / 1024, 256, 0, stream>>>(qkv_w, wqkv, n);
  }
  {
    int n = kDModel * kDModel;  // 1,048,576
    convert_f32_bf16<<<n / 1024, 256, 0, stream>>>(out_w, wo, n);
  }

  // QKV projection: [4096,1024] x [1024,3072] + bias -> bf16 qkv
  gemm_bf16_wmma<true><<<dim3(k3C / 128, kTok / 128), 256, 0, stream>>>(
      xb, wqkv, qkv_b, qkvb, kTok, k3C, kDModel);

  // Attention: 16 heads x (2 batches x 16 q-tiles)
  flash_attn_wmma<<<dim3(kSeq / 128, kHeads, kBatch), 256, 0, stream>>>(
      qkvb, ctxb);

  // Output projection: [4096,1024] x [1024,1024] + bias -> fp32 d_out
  gemm_bf16_wmma<false><<<dim3(kDModel / 128, kTok / 128), 256, 0, stream>>>(
      ctxb, wo, out_b, out, kTok, kDModel, kDModel);
}
